// JointVectorQuantizerEMA_61649960567434
// MI455X (gfx1250) — compile-verified, matrix operand order checked by reading, not run on hardware
//
#include <hip/hip_runtime.h>

// ---------------------------------------------------------------------------
// VQ-VAE EMA quantizer for MI455X (gfx1250, wave32, WMMA + async-to-LDS).
//   argmin GEMM (32768x1024x256, 17.2 GFLOP) in bf16 WMMA; codebook tiles
//   double-buffered in LDS via GLOBAL_LOAD_ASYNC_TO_LDS_B128 (ASYNCcnt).
// ---------------------------------------------------------------------------

typedef __bf16 bf16_t;
typedef __attribute__((ext_vector_type(16))) __bf16 v16bf;
typedef __attribute__((ext_vector_type(8)))  __bf16 v8bf;
typedef __attribute__((ext_vector_type(8)))  float  v8f;
typedef __attribute__((ext_vector_type(4)))  int    v4i;

// Pointer types matching the async builtin's prototype: int4 pointers in the
// device (as1) and shared (as3) address spaces.
typedef __attribute__((address_space(1))) v4i* gptr_v4i;
typedef __attribute__((address_space(3))) v4i* lptr_v4i;

#define NUM_CODES 1024
#define CODE_DIM  256
#define NPTS      32768           // B*H*W = 32*32*32
#define TOT_ELEMS 8388608.0f      // B*D*H*W
#define DECAY     0.99f
#define OMD       0.01f           // 1 - DECAY
#define EPSV      1e-5f
#define COMMIT    0.25f

// ---- CDNA5 async global->LDS copy (ASYNCcnt) ------------------------------
__device__ __forceinline__ void async_copy_b128(const void* g, void* l)
{
#if __has_builtin(__builtin_amdgcn_global_load_async_to_lds_b128)
    __builtin_amdgcn_global_load_async_to_lds_b128(
        (gptr_v4i)(uintptr_t)g, (lptr_v4i)(unsigned)(uintptr_t)l, 0, 0);
#else
    const unsigned           loff = (unsigned)(uintptr_t)l;     // LDS byte offset
    const unsigned long long ga   = (unsigned long long)(uintptr_t)g;
    asm volatile("global_load_async_to_lds_b128 %0, %1, off"
                 :: "v"(loff), "v"(ga) : "memory");
#endif
}

__device__ __forceinline__ void async_wait0()
{
#if __has_builtin(__builtin_amdgcn_s_wait_asynccnt)
    __builtin_amdgcn_s_wait_asynccnt(0);
#else
    asm volatile("s_wait_asynccnt 0x0" ::: "memory");
#endif
}

// ---------------- K0: emb -> bf16, ||e_k||^2, zero accumulators -------------
__global__ __launch_bounds__(256) void vq_prep(
    const float* __restrict__ emb, bf16_t* __restrict__ ebf,
    float* __restrict__ enorm, float* __restrict__ counts,
    float* __restrict__ embed_sum, float* __restrict__ scalars)
{
    const int k = blockIdx.x;
    const int t = threadIdx.x;
    const size_t i = (size_t)k * CODE_DIM + t;
    const float v = emb[i];
    ebf[i] = (bf16_t)v;
    embed_sum[i] = 0.0f;
    if (t == 0) counts[k] = 0.0f;
    if (k == 0 && t < 8) scalars[t] = 0.0f;

    float p = v * v;
#pragma unroll
    for (int m = 16; m >= 1; m >>= 1) p += __shfl_xor(p, m, 32);
    __shared__ float sred[8];
    if ((t & 31) == 0) sred[t >> 5] = p;
    __syncthreads();
    if (t == 0) {
        float s = 0.f;
#pragma unroll
        for (int j = 0; j < 8; ++j) s += sred[j];
        enorm[k] = s;
    }
}

// ---------------- K1: WMMA nearest-code assignment --------------------------
// Block = 256 thr = 8 waves; wave owns 16 rows of x in registers as 8 bf16
// A-fragments. 32-code (16 KB) emb tiles double-buffered in LDS via async
// copies; 16x v_wmma_f32_16x16x32_bf16 per tile; running (min,argmin)/row.
__global__ __launch_bounds__(256) void vq_assign_wmma(
    const float* __restrict__ x, const bf16_t* __restrict__ ebf,
    const float* __restrict__ enorm, int* __restrict__ codes)
{
    __shared__ __align__(16) bf16_t sB[2][32 * CODE_DIM];   // 2 x 16 KB tiles

    const int tid   = threadIdx.x;
    const int lane  = tid & 31;
    const int wave  = tid >> 5;
    const int col   = lane & 15;            // N column owned in B/C layout
    const int khalf = (lane >> 4) << 3;     // 0 or 8: K sub-offset per A/B layout
    const int rowBase = blockIdx.x * 128 + wave * 16;

    // Build A fragments (ISA 16-bit A 16x32 layout): lane holds row rowBase+col;
    // elements 0..7 = K k0..k0+7, elements 8..15 = K k0+16..k0+23.
    const float* xrow = x + (size_t)(rowBase + col) * CODE_DIM;
    v16bf A[8];
#pragma unroll
    for (int kc = 0; kc < 8; ++kc) {
        const int k0 = kc * 32 + khalf;
        v8f lo = *(const v8f*)(xrow + k0);
        v8f hi = *(const v8f*)(xrow + k0 + 16);
#pragma unroll
        for (int e = 0; e < 8; ++e) {
            A[kc][e]     = (bf16_t)lo[e];
            A[kc][8 + e] = (bf16_t)hi[e];
        }
    }

    float bestV[8];
    int   bestI[8];
#pragma unroll
    for (int j = 0; j < 8; ++j) { bestV[j] = 3.4e38f; bestI[j] = 0; }

    // Async stage of one 16 KB tile: 256 threads x 4 x b128 (16 B per lane).
    auto stage = [&](int buf, int tile) {
        const char* gp = (const char*)(ebf + (size_t)tile * 32 * CODE_DIM) + tid * 16;
        char*       lp = (char*)&sB[buf][0] + tid * 16;
#pragma unroll
        for (int p = 0; p < 4; ++p)
            async_copy_b128(gp + p * 4096, lp + p * 4096);
    };

    const int NTILE = NUM_CODES / 32;       // 32 tiles
    stage(0, 0);                            // prologue

    for (int tile = 0; tile < NTILE; ++tile) {
        const int buf = tile & 1;
        async_wait0();                      // my wave's copies landed in LDS
        __syncthreads();                    // everyone's copies landed
        if (tile + 1 < NTILE) stage(buf ^ 1, tile + 1);   // prefetch next tile

#pragma unroll
        for (int cb = 0; cb < 2; ++cb) {
            v8f c = {0.f, 0.f, 0.f, 0.f, 0.f, 0.f, 0.f, 0.f};
            const bf16_t* brow = &sB[buf][0] + ((cb * 16 + col) << 8);
#pragma unroll
            for (int kc = 0; kc < 8; ++kc) {
                const int k0 = kc * 32 + khalf;
                v8bf blo = *(const v8bf*)(brow + k0);
                v8bf bhi = *(const v8bf*)(brow + k0 + 16);
                v16bf bb = __builtin_shufflevector(blo, bhi,
                    0, 1, 2, 3, 4, 5, 6, 7, 8, 9, 10, 11, 12, 13, 14, 15);
                c = __builtin_amdgcn_wmma_f32_16x16x32_bf16(
                        false, A[kc], false, bb, (short)0, c, false, false);
            }
            const int   codeIdx = tile * 32 + cb * 16 + col;
            const float en = enorm[codeIdx];            // 4 KB, L2-resident
#pragma unroll
            for (int j = 0; j < 8; ++j) {
                const float score = en - 2.0f * c[j];   // ||e||^2 - 2 x.e
                if (score < bestV[j] || (score == bestV[j] && codeIdx < bestI[j])) {
                    bestV[j] = score; bestI[j] = codeIdx;
                }
            }
        }
        __syncthreads();   // all waves done with buf before it is refilled (tile+2)
    }

    // Reduce (min,idx) across the 16 lanes sharing each row (wave32 halves).
#pragma unroll
    for (int j = 0; j < 8; ++j) {
        float v = bestV[j];
        int idx = bestI[j];
#pragma unroll
        for (int m = 8; m >= 1; m >>= 1) {
            const float ov = __shfl_xor(v, m, 16);
            const int   oi = __shfl_xor(idx, m, 16);
            if (ov < v || (ov == v && oi < idx)) { v = ov; idx = oi; }
        }
        if (col == 0) codes[rowBase + khalf + j] = idx;  // lane0: rows j, lane16: rows j+8
    }
}

// ---------------- K2: gather x_q, loss, counts, embed_sum -------------------
__global__ __launch_bounds__(256) void vq_gather_update(
    const float* __restrict__ x, const float* __restrict__ emb,
    const int* __restrict__ codes, float* __restrict__ out_xq,
    float* __restrict__ out_codes, float* __restrict__ counts,
    float* __restrict__ embed_sum, float* __restrict__ scalars)
{
    const int bh = blockIdx.x;          // b*32 + h
    const int b = bh >> 5, h = bh & 31;
    const int t = threadIdx.x;
    __shared__ int sCode[32];
    if (t < 32) {
        const int c = codes[b * 1024 + h * 32 + t];
        sCode[t] = c;
        out_codes[b * 1024 + h * 32 + t] = (float)c;
        atomicAdd(&counts[c], 1.0f);
    }
    __syncthreads();

    const int w = t & 31, dl = t >> 5;
    const int cw = sCode[w];
    float acc = 0.f;
    for (int d0 = 0; d0 < CODE_DIM; d0 += 8) {
        const int d = d0 + dl;
        const size_t idx = (size_t)b * 262144 + (size_t)d * 1024 + h * 32 + w;
        const float xv = x[idx];
        const float q = emb[(size_t)cw * CODE_DIM + d];
        out_xq[idx] = q;                 // x + sg(x_q - x) == x_q numerically
        const float diff = q - xv;
        acc += diff * diff;
        atomicAdd(&embed_sum[(size_t)cw * CODE_DIM + d], xv);
    }
#pragma unroll
    for (int m = 16; m >= 1; m >>= 1) acc += __shfl_xor(acc, m, 32);
    __shared__ float sred[8];
    if ((t & 31) == 0) sred[t >> 5] = acc;
    __syncthreads();
    if (t == 0) {
        float s = 0.f;
#pragma unroll
        for (int i = 0; i < 8; ++i) s += sred[i];
        atomicAdd(&scalars[0], s);
    }
}

// ---------------- K3a: cluster-size EMA, n, loss scalar ---------------------
__global__ __launch_bounds__(1024) void vq_finalize_small(
    const float* __restrict__ cluster_size, const float* __restrict__ counts,
    float* __restrict__ ncs_ws, float* __restrict__ out_ncs,
    float* __restrict__ out_loss, float* __restrict__ scalars)
{
    const int k = threadIdx.x;
    const float ncs = DECAY * cluster_size[k] + OMD * counts[k];
    ncs_ws[k] = ncs;
    out_ncs[k] = ncs;
    float p = ncs;
#pragma unroll
    for (int m = 16; m >= 1; m >>= 1) p += __shfl_xor(p, m, 32);
    __shared__ float sred[32];
    if ((k & 31) == 0) sred[k >> 5] = p;
    __syncthreads();
    if (k == 0) {
        float n = 0.f;
#pragma unroll
        for (int i = 0; i < 32; ++i) n += sred[i];
        scalars[1] = n;
        out_loss[0] = scalars[0] * (1.0f + COMMIT) / TOT_ELEMS;
    }
}

// ---------------- K3b: embed_avg EMA + normalization ------------------------
__global__ __launch_bounds__(256) void vq_finalize_embed(
    const float* __restrict__ embed_avg, const float* __restrict__ embed_sum,
    const float* __restrict__ ncs_ws, const float* __restrict__ scalars,
    float* __restrict__ out_nea, float* __restrict__ out_en)
{
    const int k = blockIdx.x, d = threadIdx.x;
    const size_t i = (size_t)k * CODE_DIM + d;
    const float nea = DECAY * embed_avg[i] + OMD * embed_sum[i];
    out_nea[i] = nea;
    const float n = scalars[1];
    const float cs = (ncs_ws[k] + EPSV) / (n + NUM_CODES * EPSV) * n;
    out_en[i] = nea / cs;
}

// ---------------------------------------------------------------------------
extern "C" void kernel_launch(void* const* d_in, const int* in_sizes, int n_in,
                              void* d_out, int out_size, void* d_ws, size_t ws_size,
                              hipStream_t stream)
{
    const float* x            = (const float*)d_in[0];  // [32,256,32,32]
    const float* emb          = (const float*)d_in[1];  // [1024,256]
    const float* cluster_size = (const float*)d_in[2];  // [1024]
    const float* embed_avg    = (const float*)d_in[3];  // [1024,256]
    float* out = (float*)d_out;

    // Workspace layout (~1.7 MB)
    char* ws = (char*)d_ws;
    bf16_t* ebf       = (bf16_t*)(ws);                  // 512 KB bf16 codebook
    float*  enorm     = (float*)(ws + (512 << 10));     //   4 KB ||e||^2
    float*  counts    = (float*)(ws + (516 << 10));     //   4 KB
    float*  ncs       = (float*)(ws + (520 << 10));     //   4 KB
    float*  scalars   = (float*)(ws + (524 << 10));     //  [0]=loss acc, [1]=n
    int*    codes     = (int*)  (ws + (528 << 10));     // 128 KB
    float*  embed_sum = (float*)(ws + (672 << 10));     //   1 MB

    // d_out flat concat, return order:
    float* out_xq    = out;                 // 8388608
    float* out_loss  = out + 8388608;       // 1
    float* out_codes = out + 8388609;       // 32768
    float* out_ncs   = out + 8421377;       // 1024
    float* out_nea   = out + 8422401;       // 262144
    float* out_en    = out + 8684545;       // 262144

    vq_prep<<<NUM_CODES, 256, 0, stream>>>(emb, ebf, enorm, counts, embed_sum, scalars);
    vq_assign_wmma<<<NPTS / 128, 256, 0, stream>>>(x, ebf, enorm, codes);
    vq_gather_update<<<1024, 256, 0, stream>>>(x, emb, codes, out_xq, out_codes,
                                               counts, embed_sum, scalars);
    vq_finalize_small<<<1, 1024, 0, stream>>>(cluster_size, counts, ncs, out_ncs,
                                              out_loss, scalars);
    vq_finalize_embed<<<NUM_CODES, 256, 0, stream>>>(embed_avg, embed_sum, ncs,
                                                     scalars, out_nea, out_en);
}
